// BaselineOrbitals_42374147342458
// MI455X (gfx1250) — matrix operand order repaired and checked by reading, not compiled, not run on hardware
//
#include <hip/hip_runtime.h>
#include <hip/hip_bf16.h>
#include <math.h>

typedef __bf16 bf16_t;
typedef __attribute__((ext_vector_type(16))) __bf16 v16bf;
typedef __attribute__((ext_vector_type(8)))  __bf16 v8bf;
typedef __attribute__((ext_vector_type(4)))  __bf16 v4bf;
typedef __attribute__((ext_vector_type(8)))  float  v8f;
typedef __attribute__((ext_vector_type(4)))  float  v4f;
typedef __attribute__((ext_vector_type(4)))  unsigned int u32x4;
typedef __attribute__((ext_vector_type(8)))  int    i32x8;
typedef __attribute__((ext_vector_type(4)))  int    i32x4;

namespace {
constexpr int kB      = 384;
constexpr int kNEl    = 32;
constexpr int kNIon   = 8;
constexpr int kD      = 256;     // electron feature dim
constexpr int kP      = 32;      // pair embedding dim
constexpr int kH      = 256;     // hidden
constexpr int kDP     = kD + kP; // 288
constexpr int kNDets  = 16;
constexpr int kNOrb   = 32;      // N_UP + N_DN
constexpr int kNMo    = 64;
constexpr int kNBasis = 14;
constexpr int kNAo    = 112;
constexpr int kKAo    = 128;     // padded K for MO GEMM
}

// ---------- scalar helpers ----------

__device__ __forceinline__ bf16_t f2bf(float f) { return (bf16_t)f; }  // native cvt on gfx1250

__device__ __forceinline__ float fast_tanh(float x) {
#if __has_builtin(__builtin_amdgcn_tanhf)
  return __builtin_amdgcn_tanhf(x);       // v_tanh_f32
#else
  float e = __expf(2.0f * x);
  return 1.0f - 2.0f / (e + 1.0f);
#endif
}

__device__ __forceinline__ v8f vzero8() {
  v8f c;
#pragma unroll
  for (int e = 0; e < 8; ++e) c[e] = 0.0f;
  return c;
}

// ---------- CDNA5 async global->LDS copies (ASYNCcnt) ----------

__device__ __forceinline__ unsigned lds_addr(void* p) {
  // flat shared pointer: LDS offset lives in addr[31:0]
  return (unsigned)(uintptr_t)p;
}

__device__ __forceinline__ void async_copy_b128(const bf16_t* g, bf16_t* l) {
  asm volatile("global_load_async_to_lds_b128 %0, %1, off"
               :: "v"(lds_addr(l)), "v"((unsigned long long)(uintptr_t)g)
               : "memory");
}

__device__ __forceinline__ void wait_async(int n) {
  if (n == 0) asm volatile("s_wait_asynccnt 0x0" ::: "memory");
  else        asm volatile("s_wait_asynccnt 0x1" ::: "memory");
}

// ---------- CDNA5 Tensor Data Mover (TENSORcnt) ----------

#if __has_builtin(__builtin_amdgcn_tensor_load_to_lds)
#define HAVE_TDM 1
#else
#define HAVE_TDM 0
#endif

__device__ __forceinline__ void tdm_wait() {
#if __has_builtin(__builtin_amdgcn_s_wait_tensorcnt)
  __builtin_amdgcn_s_wait_tensorcnt((short)0);
#else
  asm volatile("s_wait_tensorcnt 0x0" ::: "memory");
#endif
}

#if HAVE_TDM
// 1-D contiguous tile DMA: nbytes (multiple of 8) from global to LDS.
// D# group0: count=1, lds_addr, global_addr, type=2; group1: data_size=8B,
// tensor_dim0 = tile_dim0 = nbytes/8, stride0 = dim0, no padding/iterate/multicast.
__device__ __forceinline__ void tdm_load_1d(const void* gsrc, void* ldst, unsigned nbytes) {
  unsigned n = nbytes >> 3;
  unsigned long long ga = (unsigned long long)(uintptr_t)gsrc;
  u32x4 g0;
  g0[0] = 1u;                                    // count=1, user mode
  g0[1] = lds_addr(ldst);                        // lds byte address
  g0[2] = (unsigned)ga;                          // global_addr[31:0]
  g0[3] = ((unsigned)(ga >> 32) & 0x01FFFFFFu) | (2u << 30);  // addr[56:32] | type=2
  i32x8 g1;
  g1[0] = (int)(3u << 16);                       // data_size = 8 bytes
  g1[1] = (int)((n & 0xFFFFu) << 16);            // tensor_dim0[15:0] (bits 63:48)
  g1[2] = (int)((n >> 16) & 0xFFFFu);            // tensor_dim0[31:16]; tensor_dim1=0
  g1[3] = (int)((n & 0xFFFFu) << 16);            // tile_dim0 = n (bits 127:112)
  g1[4] = 0;                                     // tile_dim1/2 unused
  g1[5] = (int)n;                                // tensor_dim0_stride[31:0]
  g1[6] = 0;
  g1[7] = 0;
  i32x4 z4; z4[0] = z4[1] = z4[2] = z4[3] = 0;
#if __clang_major__ >= 23
  i32x8 z8;
#pragma unroll
  for (int e = 0; e < 8; ++e) z8[e] = 0;
  __builtin_amdgcn_tensor_load_to_lds(g0, g1, z4, z4, z8, 0);   // therock 6-arg form
#else
  __builtin_amdgcn_tensor_load_to_lds(g0, g1, z4, z4, 0);       // ROCm 7.2 5-arg form
#endif
}
#endif

// stage a contiguous bf16 block into LDS: TDM if available (issued by wave 0),
// else per-lane async b128 copies. Caller must __syncthreads() after.
__device__ __forceinline__ void stage_block(const bf16_t* g, bf16_t* l, int nbytes,
                                            int tid, int wave) {
#if HAVE_TDM
  (void)tid;
  if (wave == 0) {
    tdm_load_1d(g, l, (unsigned)nbytes);
    tdm_wait();
  }
#else
  (void)wave;
  int nvec = nbytes / 16;
  for (int t = tid; t < nvec; t += 256) async_copy_b128(g + t * 8, l + t * 8);
  wait_async(0);
#endif
}

// ---------- WMMA fragment helpers (contiguous-k 16-bit layouts, wave32) ----------

// A fragment (16x32 bf16): lane holds row (lane&15); K = kbase + 16h + e (contiguous)
__device__ __forceinline__ v16bf a_from_bf16(const bf16_t* __restrict__ row, int kbase, int h) {
  v8bf c0 = *(const v8bf*)(row + kbase + 16 * h);
  v8bf c1 = *(const v8bf*)(row + kbase + 16 * h + 8);
  v16bf a;
#pragma unroll
  for (int e = 0; e < 8; ++e) { a[e] = c0[e]; a[8 + e] = c1[e]; }
  return a;
}

// B fragment (32x16 bf16) from N-major weight Wt[n][k]: lane holds column (lane&15),
// K = kbase + 16*(lane>>4) + e -> one contiguous 32B load
__device__ __forceinline__ v16bf b_frag(const bf16_t* __restrict__ Wt, int Kstride,
                                        int ntile, int kbase, int lane) {
  const bf16_t* p = Wt + (size_t)(ntile * 16 + (lane & 15)) * Kstride + kbase + 16 * (lane >> 4);
  return *(const v16bf*)p;
}

__device__ __forceinline__ v8f wmma_bf16(v16bf a, v16bf b, v8f c) {
  return __builtin_amdgcn_wmma_f32_16x16x32_bf16(false, a, false, b, (short)0, c, false, false);
}

// ---------- prep kernels ----------

__global__ void f32_to_bf16_vec(const float* __restrict__ in, bf16_t* __restrict__ out, int n4) {
  int i = blockIdx.x * blockDim.x + threadIdx.x;
  if (i >= n4) return;
  v4f x = *(const v4f*)(in + (size_t)i * 4);
  v4bf y;
#pragma unroll
  for (int e = 0; e < 4; ++e) y[e] = f2bf(x[e]);
  *(v4bf*)(out + (size_t)i * 4) = y;
}

__global__ void transpose_to_bf16(const float* __restrict__ in, bf16_t* __restrict__ out,
                                  int K, int N, int Kpad) {
  int idx = blockIdx.x * blockDim.x + threadIdx.x;
  int total = N * Kpad;
  if (idx >= total) return;
  int n = idx / Kpad, k = idx - n * Kpad;
  float v = (k < K) ? in[(size_t)k * N + n] : 0.0f;
  out[idx] = f2bf(v);
}

// ---------- shift MLP (rank-1 split):  base = h_el@W0[:256]+b0 shared over pairs,
//            hid = tanh(base + pair@W0[256:288]); s = hid@W1; reduce s/(1+d^3)*diff ----------

template <int NP>
__global__ void __launch_bounds__(256, 1)
shift_kernel(const bf16_t* __restrict__ hel_bf,  // (B,32,256) bf16
             const bf16_t* __restrict__ pair_bf, // (B,32,NP,32) bf16
             const float* __restrict__ diff,     // (B,32,NP,3)
             const float* __restrict__ dist,     // (B,32,NP)
             const bf16_t* __restrict__ Wt,      // (256, 288) N-major bf16
             const float* __restrict__ b0,       // (256)
             const float* __restrict__ W1,       // (256)
             float* __restrict__ s_out) {        // (B,32,3)
  constexpr int MT = (NP + 15) / 16;
  __shared__ float  base_sh[kNEl][kH + 4];
  __shared__ float  sacc[kNEl][NP];
  __shared__ bf16_t hbuf[kNEl][kD];      // staged h_el rows (16KB, TDM)
  __shared__ bf16_t pbuf[2][NP * kP];    // double-buffered pair tiles (async)

  const int b = blockIdx.x;
  const int tid = threadIdx.x;
  const int wave = tid >> 5, lane = tid & 31;
  const int h = lane >> 4, nl = lane & 15;

  // stage h_el rows (32x256 bf16 = 16KB) via TDM / async DMA
  stage_block(hel_bf + (size_t)b * kNEl * kD, &hbuf[0][0], kNEl * kD * 2, tid, wave);
  // stage pair tile for electron 0 (async; wave-aligned thread count)
  {
    const bf16_t* src = pair_bf + (size_t)b * kNEl * NP * kP;
    if (tid < NP * 4) async_copy_b128(src + tid * 8, &pbuf[0][0] + tid * 8);
  }
  for (int t = tid; t < kNEl * NP; t += 256) ((float*)sacc)[t] = 0.0f;
  wait_async(0);
  __syncthreads();

  // ---- stage 1: base GEMM (32x256 = 32x256 @ 256x256), 8 waves x 2 ntiles ----
  {
    v8f acc[2][2];
#pragma unroll
    for (int mt = 0; mt < 2; ++mt)
#pragma unroll
      for (int nt = 0; nt < 2; ++nt) acc[mt][nt] = vzero8();

#pragma unroll
    for (int ks = 0; ks < 8; ++ks) {
      __builtin_prefetch(Wt + (size_t)(2 * wave * 16) * kDP + (ks + 1) * 32, 0, 1);
      v16bf a0 = a_from_bf16(&hbuf[nl][0], ks * 32, h);
      v16bf a1 = a_from_bf16(&hbuf[16 + nl][0], ks * 32, h);
#pragma unroll
      for (int nt = 0; nt < 2; ++nt) {
        v16bf bb = b_frag(Wt, kDP, 2 * wave + nt, ks * 32, lane);
        acc[0][nt] = wmma_bf16(a0, bb, acc[0][nt]);
        acc[1][nt] = wmma_bf16(a1, bb, acc[1][nt]);
      }
    }
#pragma unroll
    for (int nt = 0; nt < 2; ++nt) {
      int col = (2 * wave + nt) * 16 + nl;
      float bias = b0[col];
#pragma unroll
      for (int mt = 0; mt < 2; ++mt)
#pragma unroll
        for (int v = 0; v < 8; ++v)
          base_sh[mt * 16 + v + 8 * h][col] = acc[mt][nt][v] + bias;
    }
  }
  __syncthreads();

  // ---- stage 2: per-electron pair GEMM (K=32) + tanh + dot(W1), async double buffer ----
  const bf16_t zb = (bf16_t)0.0f;
  float w1c[2];
#pragma unroll
  for (int nt = 0; nt < 2; ++nt) w1c[nt] = W1[(2 * wave + nt) * 16 + nl];

  const bf16_t* pair_base = pair_bf + (size_t)b * kNEl * NP * kP;
  for (int i = 0; i < kNEl; ++i) {
    if (i + 1 < kNEl) {  // prefetch next electron's pair tile
      if (tid < NP * 4)
        async_copy_b128(pair_base + (size_t)(i + 1) * NP * kP + tid * 8,
                        &pbuf[(i + 1) & 1][0] + tid * 8);
      wait_async(1);
    } else {
      wait_async(0);
    }
    __syncthreads();

    const bf16_t* pb = &pbuf[i & 1][0];
    v16bf a[MT];
#pragma unroll
    for (int mt = 0; mt < MT; ++mt) {
      int j = mt * 16 + nl;
      if (j < NP) {
        a[mt] = a_from_bf16(pb + j * kP, 0, h);
      } else {
#pragma unroll
        for (int e = 0; e < 16; ++e) a[mt][e] = zb;
      }
    }
    v8f c[MT][2];
#pragma unroll
    for (int mt = 0; mt < MT; ++mt)
#pragma unroll
      for (int nt = 0; nt < 2; ++nt) c[mt][nt] = vzero8();

#pragma unroll
    for (int nt = 0; nt < 2; ++nt) {
      v16bf bb = b_frag(Wt, kDP, 2 * wave + nt, kD, lane);  // K rows 256..287
#pragma unroll
      for (int mt = 0; mt < MT; ++mt) c[mt][nt] = wmma_bf16(a[mt], bb, c[mt][nt]);
    }

    float part[MT][8];
#pragma unroll
    for (int mt = 0; mt < MT; ++mt)
#pragma unroll
      for (int v = 0; v < 8; ++v) part[mt][v] = 0.0f;

#pragma unroll
    for (int nt = 0; nt < 2; ++nt) {
      int col = (2 * wave + nt) * 16 + nl;
      float bval = base_sh[i][col];
#pragma unroll
      for (int mt = 0; mt < MT; ++mt)
#pragma unroll
        for (int v = 0; v < 8; ++v) {
          int j = mt * 16 + v + 8 * h;
          if (j < NP) part[mt][v] += fast_tanh(bval + c[mt][nt][v]) * w1c[nt];
        }
    }
    // reduce over the 16 lanes of each half-wave (sum over hidden columns)
#pragma unroll
    for (int off = 8; off >= 1; off >>= 1)
#pragma unroll
      for (int mt = 0; mt < MT; ++mt)
#pragma unroll
        for (int v = 0; v < 8; ++v) part[mt][v] += __shfl_xor(part[mt][v], off, 32);

    if (nl == 0) {
#pragma unroll
      for (int mt = 0; mt < MT; ++mt)
#pragma unroll
        for (int v = 0; v < 8; ++v) {
          int j = mt * 16 + v + 8 * h;
          if (j < NP) atomicAdd(&sacc[i][j], part[mt][v]);
        }
    }
    __syncthreads();
  }

  // ---- stage 3: shift = sum_j (s/(1+d^3)) * diff ----
  {
    int i = tid >> 3, l8 = tid & 7;
    float a0 = 0.f, a1 = 0.f, a2 = 0.f;
    for (int j = l8; j < NP; j += 8) {
      float d = dist[(size_t)(b * kNEl + i) * NP + j];
      float w = sacc[i][j] / (1.0f + d * d * d);
      const float* dp = diff + ((size_t)(b * kNEl + i) * NP + j) * 3;
      a0 += w * dp[0]; a1 += w * dp[1]; a2 += w * dp[2];
    }
#pragma unroll
    for (int off = 4; off >= 1; off >>= 1) {
      a0 += __shfl_xor(a0, off, 32);
      a1 += __shfl_xor(a1, off, 32);
      a2 += __shfl_xor(a2, off, 32);
    }
    if (l8 == 0) {
      float* o = s_out + (size_t)(b * kNEl + i) * 3;
      o[0] = a0; o[1] = a1; o[2] = a2;
    }
  }
}

// ---------- backflow hidden: tanh(h_el[:,spin*16+..]@W_bf0 + b0) -> bf16 ----------

__global__ void __launch_bounds__(256, 1)
bf_hidden_kernel(const bf16_t* __restrict__ hel_bf,
                 const bf16_t* __restrict__ Wt_up0, const float* __restrict__ b_up0,
                 const bf16_t* __restrict__ Wt_dn0, const float* __restrict__ b_dn0,
                 bf16_t* __restrict__ bfh) {  // (2, B, 16, 256) bf16
  __shared__ bf16_t hbuf[kNEl][kD];  // 16KB
  const int b = blockIdx.x, tid = threadIdx.x;
  const int wave = tid >> 5, lane = tid & 31, h = lane >> 4, nl = lane & 15;
  const int spin = wave >> 2, wl = wave & 3;
  const bf16_t* Wt = spin ? Wt_dn0 : Wt_up0;
  const float* bias = spin ? b_dn0 : b_up0;

  stage_block(hel_bf + (size_t)b * kNEl * kD, &hbuf[0][0], kNEl * kD * 2, tid, wave);
  __syncthreads();

  const bf16_t* A = &hbuf[spin * 16 + nl][0];
  v8f acc[4];
#pragma unroll
  for (int nt = 0; nt < 4; ++nt) acc[nt] = vzero8();
#pragma unroll
  for (int ks = 0; ks < 8; ++ks) {
    v16bf a = a_from_bf16(A, ks * 32, h);
#pragma unroll
    for (int nt = 0; nt < 4; ++nt) {
      v16bf bb = b_frag(Wt, kD, wl * 4 + nt, ks * 32, lane);
      acc[nt] = wmma_bf16(a, bb, acc[nt]);
    }
  }
#pragma unroll
  for (int nt = 0; nt < 4; ++nt) {
    int col = (wl * 4 + nt) * 16 + nl;
    float bs = bias[col];
#pragma unroll
    for (int v = 0; v < 8; ++v) {
      int m = v + 8 * h;
      bfh[((size_t)(spin * kB + b) * 16 + m) * kD + col] = f2bf(fast_tanh(acc[nt][v] + bs));
    }
  }
}

// ---------- final: decay/shift, AO, MO GEMM, gather+CI, backflow GEMM, write ----------

__global__ void __launch_bounds__(256, 1)
final_kernel(const float* __restrict__ h_ion,        // (8,64)
             const float* __restrict__ diff_el_ion,  // (B,32,8,3)
             const float* __restrict__ dist_el_ion,  // (B,32,8)
             const float* __restrict__ decay_scale,  // (1)
             const float* __restrict__ W_decay,      // (64)
             const float* __restrict__ b_decay,      // (1)
             const float* __restrict__ alpha,        // (14)
             const float* __restrict__ ci,           // (16)
             const int* __restrict__ idx_up,         // (16,16)
             const int* __restrict__ idx_dn,         // (16,16)
             const float* __restrict__ b_bf_up1,     // (512)
             const float* __restrict__ b_bf_dn1,     // (512)
             const float* __restrict__ s_el,         // (B,32,3)
             const float* __restrict__ s_ion,        // (B,32,3)
             const bf16_t* __restrict__ Wt_mo_up,    // (64,128)
             const bf16_t* __restrict__ Wt_mo_dn,
             const bf16_t* __restrict__ Wt_bf1_up,   // (512,256)
             const bf16_t* __restrict__ Wt_bf1_dn,
             const bf16_t* __restrict__ bfh,         // (2,B,16,256)
             float* __restrict__ out) {
  __shared__ float  ls_sh[kNIon];
  __shared__ float  shift_sh[kNEl][3];
  __shared__ bf16_t ao_sh[kNEl][kKAo];
  __shared__ float  mo_sh[2][16][kNMo];
  __shared__ float  ciw_sh[kNDets];
  __shared__ float  sgn_sh[kNDets];
  __shared__ int    idx_sh[2][kNDets][16];
  __shared__ bf16_t abuf[16 * kD];  // staged backflow-hidden A block (8KB)

  const int b = blockIdx.x, tid = threadIdx.x;

  if (tid < kNIon) {
    float acc = 0.f;
#pragma unroll
    for (int k = 0; k < 64; ++k) acc += h_ion[tid * 64 + k] * W_decay[k];
    ls_sh[tid] = decay_scale[0] / fast_tanh(acc + b_decay[0]);
  }
  if (tid < kNDets) {
    float c = ci[tid];
    ciw_sh[tid] = powf(fabsf(c), 1.0f / 16.0f);
    sgn_sh[tid] = (c > 0.f) ? 1.f : ((c < 0.f) ? -1.f : 0.f);
  }
  for (int t = tid; t < 2 * kNDets * 16; t += 256)
    ((int*)idx_sh)[t] = (t < kNDets * 16) ? idx_up[t] : idx_dn[t - kNDets * 16];
  __syncthreads();

  if (tid < kNEl) {
    int i = tid;
    float dec = 1.f;
#pragma unroll
    for (int I = 0; I < kNIon; ++I) {
      float dv = dist_el_ion[(size_t)(b * kNEl + i) * kNIon + I] / ls_sh[I];
      dec *= fast_tanh(dv * dv);
    }
#pragma unroll
    for (int c = 0; c < 3; ++c) {
      size_t ix = (size_t)(b * kNEl + i) * 3 + c;
      shift_sh[i][c] = (s_el[ix] + s_ion[ix]) * dec;
    }
  }
  __syncthreads();

  {  // dist^2 after shift + AO values (bf16, K padded to 128)
    int i = tid >> 3, I = tid & 7;
    const float* dp = diff_el_ion + ((size_t)(b * kNEl + i) * kNIon + I) * 3;
    float dx = dp[0] + shift_sh[i][0];
    float dy = dp[1] + shift_sh[i][1];
    float dz = dp[2] + shift_sh[i][2];
    float d2 = dx * dx + dy * dy + dz * dz;
#pragma unroll
    for (int q = 0; q < kNBasis; ++q)
      ao_sh[i][I * kNBasis + q] = f2bf(__expf(-d2 * alpha[q]));
    if (I == 7) {
      for (int k = kNAo; k < kKAo; ++k) ao_sh[i][k] = (bf16_t)0.0f;
    }
  }
  __syncthreads();

  {  // MO GEMM: per spin 16x64 = 16x128 @ 128x64
    int wave = tid >> 5, lane = tid & 31, h = lane >> 4, nl = lane & 15;
    int spin = wave >> 2, nt = wave & 3;
    const bf16_t* Wm = spin ? Wt_mo_dn : Wt_mo_up;
    const bf16_t* Arow = &ao_sh[spin * 16 + nl][0];
    v8f acc = vzero8();
#pragma unroll
    for (int ks = 0; ks < 4; ++ks) {
      v16bf a = a_from_bf16(Arow, ks * 32, h);
      v16bf bb = b_frag(Wm, kKAo, nt, ks * 32, lane);
      acc = wmma_bf16(a, bb, acc);
    }
#pragma unroll
    for (int v = 0; v < 8; ++v) mo_sh[spin][v + 8 * h][nt * 16 + nl] = acc[v];
  }
  __syncthreads();

  {  // backflow GEMM (16x512 = 16x256 @ 256x512) fused with gather/CI/output
    int wave = tid >> 5, lane = tid & 31, h = lane >> 4, nl = lane & 15;
    const size_t half = (size_t)kB * kNDets * 16 * kNOrb;
    for (int spin = 0; spin < 2; ++spin) {
      const bf16_t* Wt = spin ? Wt_bf1_dn : Wt_bf1_up;
      const float* b1 = spin ? b_bf_dn1 : b_bf_up1;
      // stage this spin's A block (16x256 bf16 = 8KB) via TDM / async DMA
      stage_block(bfh + (size_t)(spin * kB + b) * 16 * kD, abuf, 16 * kD * 2, tid, wave);
      __syncthreads();

      const bf16_t* A = abuf + (size_t)nl * kD;
      v8f acc[4];
#pragma unroll
      for (int nt = 0; nt < 4; ++nt) acc[nt] = vzero8();
#pragma unroll
      for (int ks = 0; ks < 8; ++ks) {
        __builtin_prefetch(Wt + (size_t)(wave * 4 * 16) * kD + (ks + 1) * 32, 0, 1);
        v16bf a = a_from_bf16(A, ks * 32, h);
#pragma unroll
        for (int nt = 0; nt < 4; ++nt) {
          v16bf bb = b_frag(Wt, kD, wave * 4 + nt, ks * 32, lane);
          acc[nt] = wmma_bf16(a, bb, acc[nt]);
        }
      }
#pragma unroll
      for (int nt = 0; nt < 4; ++nt) {
        int col = (wave * 4 + nt) * 16 + nl;
        int det = col >> 5, orb = col & 31;
        float bias = b1[col];
#pragma unroll
        for (int v = 0; v < 8; ++v) {
          int i = v + 8 * h;
          float y = acc[nt][v] + bias;
          float fac;
          if (spin == 0) {
            fac = (orb < 16) ? mo_sh[0][i][idx_sh[0][det][orb]] * ciw_sh[det] : 0.0f;
            if (orb == 0) fac *= sgn_sh[det];
          } else {
            fac = (orb >= 16) ? mo_sh[1][i][idx_sh[1][det][orb - 16]] : 0.0f;
          }
          out[spin * half + (((size_t)b * kNDets + det) * 16 + i) * kNOrb + orb] = fac * y;
        }
      }
      __syncthreads();
    }
  }
}

// ---------- host launch ----------

extern "C" void kernel_launch(void* const* d_in, const int* in_sizes, int n_in,
                              void* d_out, int out_size, void* d_ws, size_t ws_size,
                              hipStream_t stream) {
  (void)in_sizes; (void)n_in; (void)out_size; (void)ws_size;

  const float* h_el        = (const float*)d_in[0];
  const float* h_el_el     = (const float*)d_in[1];
  const float* h_el_ion    = (const float*)d_in[2];
  const float* h_ion       = (const float*)d_in[3];
  const float* diff_el_el  = (const float*)d_in[4];
  const float* dist_el_el  = (const float*)d_in[5];
  const float* diff_el_ion = (const float*)d_in[6];
  const float* dist_el_ion = (const float*)d_in[7];
  const float* W_shift_el0 = (const float*)d_in[8];
  const float* b_shift_el0 = (const float*)d_in[9];
  const float* W_shift_el1 = (const float*)d_in[10];
  const float* W_shift_ion0= (const float*)d_in[11];
  const float* b_shift_ion0= (const float*)d_in[12];
  const float* W_shift_ion1= (const float*)d_in[13];
  const float* decay_scale = (const float*)d_in[14];
  const float* W_decay     = (const float*)d_in[15];
  const float* b_decay     = (const float*)d_in[16];
  const float* alpha       = (const float*)d_in[17];
  const float* mo_coeff_up = (const float*)d_in[18];
  const float* mo_coeff_dn = (const float*)d_in[19];
  const float* ci_weights  = (const float*)d_in[20];
  const float* W_bf_up0    = (const float*)d_in[21];
  const float* b_bf_up0    = (const float*)d_in[22];
  const float* W_bf_up1    = (const float*)d_in[23];
  const float* b_bf_up1    = (const float*)d_in[24];
  const float* W_bf_dn0    = (const float*)d_in[25];
  const float* b_bf_dn0    = (const float*)d_in[26];
  const float* W_bf_dn1    = (const float*)d_in[27];
  const float* b_bf_dn1    = (const float*)d_in[28];
  const int*   idx_up      = (const int*)d_in[29];
  const int*   idx_dn      = (const int*)d_in[30];

  char* ws = (char*)d_ws;
  size_t off = 0;
  auto take = [&](size_t bytes) -> char* {
    char* p = ws + off;
    off = (off + bytes + 255) & ~(size_t)255;
    return p;
  };
  bf16_t* wt_sel   = (bf16_t*)take((size_t)kH * kDP * 2);       // (256,288)
  bf16_t* wt_sion  = (bf16_t*)take((size_t)kH * kDP * 2);
  bf16_t* wt_up0   = (bf16_t*)take((size_t)kH * kD * 2);        // (256,256)
  bf16_t* wt_dn0   = (bf16_t*)take((size_t)kH * kD * 2);
  bf16_t* wt_up1   = (bf16_t*)take((size_t)512 * kH * 2);       // (512,256)
  bf16_t* wt_dn1   = (bf16_t*)take((size_t)512 * kH * 2);
  bf16_t* wt_moup  = (bf16_t*)take((size_t)kNMo * kKAo * 2);    // (64,128)
  bf16_t* wt_modn  = (bf16_t*)take((size_t)kNMo * kKAo * 2);
  float*  s_el_ws  = (float*)take((size_t)kB * kNEl * 3 * 4);
  float*  s_ion_ws = (float*)take((size_t)kB * kNEl * 3 * 4);
  bf16_t* bfh_ws   = (bf16_t*)take((size_t)2 * kB * 16 * kD * 2);
  bf16_t* hel_bf   = (bf16_t*)take((size_t)kB * kNEl * kD * 2);
  bf16_t* hee_bf   = (bf16_t*)take((size_t)kB * kNEl * kNEl * kP * 2);
  bf16_t* hei_bf   = (bf16_t*)take((size_t)kB * kNEl * kNIon * kP * 2);

  auto tgrid = [](int n) { return dim3((n + 255) / 256); };

  // activation conversions (vectorized x4)
  {
    int n_hel = kB * kNEl * kD / 4;
    int n_hee = kB * kNEl * kNEl * kP / 4;
    int n_hei = kB * kNEl * kNIon * kP / 4;
    f32_to_bf16_vec<<<tgrid(n_hel), 256, 0, stream>>>(h_el, hel_bf, n_hel);
    f32_to_bf16_vec<<<tgrid(n_hee), 256, 0, stream>>>(h_el_el, hee_bf, n_hee);
    f32_to_bf16_vec<<<tgrid(n_hei), 256, 0, stream>>>(h_el_ion, hei_bf, n_hei);
  }

  transpose_to_bf16<<<tgrid(kH * kDP), 256, 0, stream>>>(W_shift_el0,  wt_sel,  kDP, kH,  kDP);
  transpose_to_bf16<<<tgrid(kH * kDP), 256, 0, stream>>>(W_shift_ion0, wt_sion, kDP, kH,  kDP);
  transpose_to_bf16<<<tgrid(kH * kD),  256, 0, stream>>>(W_bf_up0,     wt_up0,  kD,  kH,  kD);
  transpose_to_bf16<<<tgrid(kH * kD),  256, 0, stream>>>(W_bf_dn0,     wt_dn0,  kD,  kH,  kD);
  transpose_to_bf16<<<tgrid(512 * kH), 256, 0, stream>>>(W_bf_up1,     wt_up1,  kH,  512, kH);
  transpose_to_bf16<<<tgrid(512 * kH), 256, 0, stream>>>(W_bf_dn1,     wt_dn1,  kH,  512, kH);
  transpose_to_bf16<<<tgrid(kNMo * kKAo), 256, 0, stream>>>(mo_coeff_up, wt_moup, kNAo, kNMo, kKAo);
  transpose_to_bf16<<<tgrid(kNMo * kKAo), 256, 0, stream>>>(mo_coeff_dn, wt_modn, kNAo, kNMo, kKAo);

  shift_kernel<32><<<kB, 256, 0, stream>>>(hel_bf, hee_bf, diff_el_el, dist_el_el,
                                           wt_sel, b_shift_el0, W_shift_el1, s_el_ws);
  shift_kernel<8><<<kB, 256, 0, stream>>>(hel_bf, hei_bf, diff_el_ion, dist_el_ion,
                                          wt_sion, b_shift_ion0, W_shift_ion1, s_ion_ws);
  bf_hidden_kernel<<<kB, 256, 0, stream>>>(hel_bf, wt_up0, b_bf_up0, wt_dn0, b_bf_dn0, bfh_ws);

  final_kernel<<<kB, 256, 0, stream>>>(h_ion, diff_el_ion, dist_el_ion, decay_scale,
                                       W_decay, b_decay, alpha, ci_weights, idx_up, idx_dn,
                                       b_bf_up1, b_bf_dn1, s_el_ws, s_ion_ws,
                                       wt_moup, wt_modn, wt_up1, wt_dn1, bfh_ws,
                                       (float*)d_out);
}
